// MultiscaleDeformableAttention_77163382440592
// MI455X (gfx1250) — matrix operand order repaired and checked by reading, not compiled, not run on hardware
//
#include <hip/hip_runtime.h>

#define BATCH 8
#define NQ    900
#define NIMG  21760
#define CDIM  256
#define NHEAD 8
#define NLVL  4
#define NPNT  4

typedef __attribute__((ext_vector_type(16))) __bf16 v16bf;
typedef __attribute__((ext_vector_type(8)))  __bf16 v8bf;
typedef __attribute__((ext_vector_type(8)))  float  v8f;
typedef __attribute__((ext_vector_type(4)))  float  v4f;

// ---------------------------------------------------------------------------
// Transpose + convert both 256x256 weight matrices to bf16 [N][K] layout so
// WMMA B-fragments become contiguous 16B loads.
// ---------------------------------------------------------------------------
__global__ void convert_weights_kernel(const float* __restrict__ Wimg,
                                       const float* __restrict__ Wout,
                                       __bf16* __restrict__ WtImg,
                                       __bf16* __restrict__ WtOut) {
  int n = blockIdx.x;        // output column (row of transposed matrix)
  int k = threadIdx.x;       // reduction index
  WtImg[(size_t)n * CDIM + k] = (__bf16)Wimg[(size_t)k * CDIM + n];
  WtOut[(size_t)n * CDIM + k] = (__bf16)Wout[(size_t)k * CDIM + n];
}

// ---------------------------------------------------------------------------
// Wave-level WMMA GEMM with 1x4 N-blocking:
//   C[M,N] = A[M,K] (f32, cvt->bf16 in-register) * Bt[N,K] (bf16) + bias[N]
// Each wave computes a 16x64 strip (4 accumulators); the A-fragment (the
// expensive f32 side) is loaded once per k-step and reused for 4 WMMAs.
// K = N = 256.  8 waves / block.
//
// A-frag layout (16-bit A 16x32, ISA 7.12.2): lane l -> row l%16,
//   element j -> K = 16*(j/8) + 8*(l/16) + (j%8)   (two 8-float runs)
// B-frag layout (from ISA B 64x16 pattern): lane l -> col l%16,
//   element j -> K = 16*(j/8) + 8*(l/16) + (j%8)                 (two 8-bf16 runs in Bt)
// C/D layout: VGPR r -> row r + 8*(l/16), col l%16.
// ---------------------------------------------------------------------------
__global__ void wmma_gemm_bias_kernel(const float* __restrict__ A,
                                      const __bf16* __restrict__ Bt,
                                      const float* __restrict__ bias,
                                      const unsigned char* __restrict__ mask,
                                      float* __restrict__ outF,
                                      __bf16* __restrict__ outBf,
                                      int M) {
  const int K = CDIM, N = CDIM;
  int wave = threadIdx.x >> 5;
  int lane = threadIdx.x & 31;
  int tile = blockIdx.x * 8 + wave;
  int tM = (tile >> 2) * 16;          // N/64 == 4 strip-groups along N
  int tN = (tile & 3) * 64;
  if (tM >= M) return;

  int nr = lane & 15;                 // row (A) / col (B) within tile
  int g  = lane >> 4;                 // lane half-group

  const v8f vzero = {0.f, 0.f, 0.f, 0.f, 0.f, 0.f, 0.f, 0.f};
  v8f acc[4];
  acc[0] = vzero; acc[1] = vzero; acc[2] = vzero; acc[3] = vzero;

  const float* arow = A + (size_t)(tM + nr) * K;

#pragma unroll
  for (int kb = 0; kb < K; kb += 32) {
    // ---- A fragment: rows of A, f32 -> bf16 in-register (loaded once) ----
    const float* ap = arow + kb + g * 8;
    v4f a0 = *(const v4f*)(ap);
    v4f a1 = *(const v4f*)(ap + 4);
    v4f a2 = *(const v4f*)(ap + 16);
    v4f a3 = *(const v4f*)(ap + 20);
    v16bf af;
#pragma unroll
    for (int i = 0; i < 4; ++i) {
      af[i]      = (__bf16)a0[i];
      af[4 + i]  = (__bf16)a1[i];
      af[8 + i]  = (__bf16)a2[i];
      af[12 + i] = (__bf16)a3[i];
    }
    // ---- 4 B fragments reuse the same A fragment ----
#pragma unroll
    for (int s = 0; s < 4; ++s) {
      const v8bf* bp =
          (const v8bf*)(Bt + (size_t)(tN + 16 * s + nr) * K + kb + g * 8);
      v8bf b0 = bp[0];
      v8bf b1 = bp[2];
      v16bf bf;
#pragma unroll
      for (int i = 0; i < 8; ++i) {
        bf[i]     = b0[i];
        bf[8 + i] = b1[i];
      }
      acc[s] = __builtin_amdgcn_wmma_f32_16x16x32_bf16(
          /*neg_a=*/false, af, /*neg_b=*/false, bf,
          /*c_mod=*/(short)0, acc[s], /*reuse_a=*/false, /*reuse_b=*/false);
    }
  }

#pragma unroll
  for (int s = 0; s < 4; ++s) {
    int col = tN + 16 * s + nr;
    float bv = bias[col];
#pragma unroll
    for (int r = 0; r < 8; ++r) {
      int row = tM + r + 8 * g;
      float val = acc[s][r] + bv;
      if (mask && mask[row]) val = 0.f;        // reference: where(mask, 0, v)
      size_t idx = (size_t)row * N + col;
      if (outBf) outBf[idx] = (__bf16)val;
      else       outF[idx]  = val;
    }
  }
}

// ---------------------------------------------------------------------------
// Query-side projections (f32 precision: these feed sampling coordinates):
//   offs = q @ W_off + b_off            (256 cols)
//   attn = softmax_{L*P}(q @ W_attn+b)  (128 cols, groups of 16 per head)
// One block per (b,q) row; query row staged in LDS.
// ---------------------------------------------------------------------------
__global__ void qproj_kernel(const float* __restrict__ qf,
                             const float* __restrict__ Woff,
                             const float* __restrict__ boff,
                             const float* __restrict__ Wattn,
                             const float* __restrict__ battn,
                             float* __restrict__ offs,
                             float* __restrict__ attn) {
  __shared__ float q[CDIM];
  __shared__ float logit[NHEAD * NLVL * NPNT];  // 128
  int bq = blockIdx.x;
  int t  = threadIdx.x;
  q[t] = qf[(size_t)bq * CDIM + t];
  __syncthreads();

  float accOff = boff[t];
  for (int k = 0; k < CDIM; ++k) accOff += q[k] * Woff[k * 256 + t];
  offs[(size_t)bq * 256 + t] = accOff;

  if (t < 128) {
    float la = battn[t];
    for (int k = 0; k < CDIM; ++k) la += q[k] * Wattn[k * 128 + t];
    logit[t] = la;
  }
  __syncthreads();
  if (t < 128) {
    int base = (t >> 4) << 4;   // softmax group of 16 (l,p) per head
    float m = logit[base];
#pragma unroll
    for (int i = 1; i < 16; ++i) m = fmaxf(m, logit[base + i]);
    float s = 0.f;
#pragma unroll
    for (int i = 0; i < 16; ++i) s += __expf(logit[base + i] - m);
    attn[(size_t)bq * 128 + t] = __expf(logit[t] - m) / s;
  }
}

// ---------------------------------------------------------------------------
// Deformable sampling. One block per (b,q); wave w == head h; lane == channel.
// Gathers 64B-contiguous 32-channel bf16 vectors per corner.
// ---------------------------------------------------------------------------
__global__ void msda_sample_kernel(const __bf16* __restrict__ v,
                                   const float* __restrict__ qpts,
                                   const float* __restrict__ ivr,
                                   const long long* __restrict__ shapes,
                                   const float* __restrict__ offs,
                                   const float* __restrict__ attn,
                                   float* __restrict__ msda) {
  __shared__ float s_off[256];
  __shared__ float s_attn[128];
  __shared__ float s_qp[4];
  int bq = blockIdx.x;
  int t  = threadIdx.x;
  int b  = bq / NQ;
  s_off[t] = offs[(size_t)bq * 256 + t];
  if (t < 128) s_attn[t] = attn[(size_t)bq * 128 + t];
  if (t < 4)   s_qp[t]   = qpts[(size_t)bq * 4 + t];
  __syncthreads();

  int h = t >> 5;
  int d = t & 31;
  float acc = 0.f;
  int start = 0;
  for (int l = 0; l < NLVL; ++l) {
    int Hl = (int)shapes[l * 2 + 0];
    int Wl = (int)shapes[l * 2 + 1];
    // vr = tile(ivr[..., ::-1], 2) -> x coords scaled by ivr[...,1], y by [...,0]
    float vrx = ivr[((size_t)b * NLVL + l) * 2 + 1];
    float vry = ivr[((size_t)b * NLVL + l) * 2 + 0];
    float qp0 = s_qp[0] * vrx, qp1 = s_qp[1] * vry;
    float qp2 = s_qp[2] * vrx, qp3 = s_qp[3] * vry;
    const __bf16* vb = v + (((size_t)b * NIMG + start) * NHEAD + h) * 32 + d;
#pragma unroll
    for (int p = 0; p < NPNT; ++p) {
      float ox = s_off[h * 32 + l * 8 + p * 2 + 0];
      float oy = s_off[h * 32 + l * 8 + p * 2 + 1];
      float aw = s_attn[h * 16 + l * 4 + p];
      float x = (qp0 + ox * qp2 * 0.125f) * (float)Wl - 0.5f;  // /(2*P)=1/8
      float y = (qp1 + oy * qp3 * 0.125f) * (float)Hl - 0.5f;
      float xf = floorf(x), yf = floorf(y);
      float wx = x - xf, wy = y - yf;
      int x0 = (int)xf, y0 = (int)yf;
#pragma unroll
      for (int c = 0; c < 4; ++c) {
        int yy = y0 + (c >> 1);
        int xx = x0 + (c & 1);
        float w = ((c >> 1) ? wy : 1.f - wy) * ((c & 1) ? wx : 1.f - wx);
        bool valid = (yy >= 0) && (yy < Hl) && (xx >= 0) && (xx < Wl);
        int yc = yy < 0 ? 0 : (yy > Hl - 1 ? Hl - 1 : yy);
        int xc = xx < 0 ? 0 : (xx > Wl - 1 ? Wl - 1 : xx);
        float val = (float)vb[(size_t)(yc * Wl + xc) * (NHEAD * 32)];
        acc += aw * w * (valid ? val : 0.f);
      }
    }
    start += Hl * Wl;
  }
  // output layout (B, Q, H*d)
  msda[(size_t)bq * CDIM + h * 32 + d] = acc;
}

// ---------------------------------------------------------------------------
extern "C" void kernel_launch(void* const* d_in, const int* in_sizes, int n_in,
                              void* d_out, int out_size, void* d_ws, size_t ws_size,
                              hipStream_t stream) {
  const float*         query_feat   = (const float*)d_in[0];
  const float*         query_points = (const float*)d_in[1];
  const float*         img_feat     = (const float*)d_in[2];
  const float*         ivr          = (const float*)d_in[3];
  const unsigned char* img_mask     = (const unsigned char*)d_in[4];
  const long long*     img_shapes   = (const long long*)d_in[5];
  const float*         W_img        = (const float*)d_in[6];
  const float*         b_img        = (const float*)d_in[7];
  const float*         W_off        = (const float*)d_in[8];
  const float*         b_off        = (const float*)d_in[9];
  const float*         W_attn       = (const float*)d_in[10];
  const float*         b_attn       = (const float*)d_in[11];
  const float*         W_out        = (const float*)d_in[12];
  const float*         b_out        = (const float*)d_in[13];

  char* ws = (char*)d_ws;
  size_t off = 0;
  auto alloc = [&](size_t bytes) {
    size_t p = off;
    off += (bytes + 255) & ~(size_t)255;
    return p;
  };
  __bf16* WtImg   = (__bf16*)(ws + alloc((size_t)CDIM * CDIM * 2));
  __bf16* WtOut   = (__bf16*)(ws + alloc((size_t)CDIM * CDIM * 2));
  __bf16* vbuf    = (__bf16*)(ws + alloc((size_t)BATCH * NIMG * CDIM * 2));
  float*  offsBuf = (float*)(ws + alloc((size_t)BATCH * NQ * 256 * 4));
  float*  attnBuf = (float*)(ws + alloc((size_t)BATCH * NQ * 128 * 4));
  float*  msdaBuf = (float*)(ws + alloc((size_t)BATCH * NQ * CDIM * 4));

  // 1. weights -> bf16 transposed
  convert_weights_kernel<<<CDIM, CDIM, 0, stream>>>(W_img, W_out, WtImg, WtOut);

  // 2. value projection (dominant GEMM, bf16 WMMA, 16x64 strip per wave)
  int Mv = BATCH * NIMG;                         // 174080
  int tilesV = (Mv / 16) * (CDIM / 64);          // 43520 wave-strips
  wmma_gemm_bias_kernel<<<tilesV / 8, 256, 0, stream>>>(
      img_feat, WtImg, b_img, img_mask, nullptr, vbuf, Mv);

  // 3. offsets + attention softmax (f32)
  qproj_kernel<<<BATCH * NQ, 256, 0, stream>>>(
      query_feat, W_off, b_off, W_attn, b_attn, offsBuf, attnBuf);

  // 4. deformable bilinear sampling
  msda_sample_kernel<<<BATCH * NQ, 256, 0, stream>>>(
      vbuf, query_points, ivr, img_shapes, offsBuf, attnBuf, msdaBuf);

  // 5. output projection (bf16 WMMA, f32 output)
  int Mo = BATCH * NQ;                           // 7200
  int tilesO = (Mo / 16) * (CDIM / 64);          // 1800 wave-strips
  wmma_gemm_bias_kernel<<<tilesO / 8, 256, 0, stream>>>(
      msdaBuf, WtOut, b_out, nullptr, (float*)d_out, nullptr, Mo);
}
